// GAT_30039001268549
// MI455X (gfx1250) — compile-verified, matrix operand order checked by reading, not run on hardware
//
#include <hip/hip_runtime.h>

// ---------------------------------------------------------------------------
// 4-layer GAT for MI455X (gfx1250, wave32).
//   - GEMM h = x@W via v_wmma_f32_16x16x32_bf16 (bf16 A/B, f32 accum),
//     W pre-transposed + LDS-padded so both fragments load as B128.
//   - edge softmax + scatter-add in f32 with L2-resident atomics.
// ---------------------------------------------------------------------------

typedef __attribute__((ext_vector_type(16))) __bf16 v16bf;
typedef __attribute__((ext_vector_type(8)))  __bf16 v8bf;
typedef __attribute__((ext_vector_type(8)))  float  v8f;

#define TPB 256

// ---------------- conversion f32 -> bf16 (activations) ----------------
__global__ void k_cvt_bf16(const float* __restrict__ src, __bf16* __restrict__ dst, int n) {
  int i = blockIdx.x * blockDim.x + threadIdx.x;
  if (i < n) dst[i] = (__bf16)src[i];
}

// ---------------- W[K,64] f32 -> WT[64, K+8] bf16 (transposed, padded) ------
__global__ void k_cvt_wT(const float* __restrict__ W, __bf16* __restrict__ WT,
                         int K, int S) {
  int i = blockIdx.x * blockDim.x + threadIdx.x;
  if (i >= K * 64) return;
  int n = i / K, k = i - n * K;
  WT[n * S + k] = (__bf16)W[(size_t)k * 64 + n];
}

// ---------------- WMMA GEMM: C[N,64] = A[N,K] * W[K,64] ----------------
// One wave computes one 16x16 tile; 8 waves/block -> 32 rows x 64 cols.
// WT (transposed, row stride S=K+8 to dodge bank conflicts) staged in LDS.
template <int K>
__global__ void k_gemm_wmma(const __bf16* __restrict__ A, const __bf16* __restrict__ WT,
                            float* __restrict__ Cout, int nrows) {
  constexpr int S = K + 8;                 // padded stride (multiple of 8 -> 16B align)
  __shared__ __bf16 sWt[64 * S];
  int tid = threadIdx.x;
  {
    const uint4* src = (const uint4*)WT;   // copy padded array as 16B chunks
    uint4*       dst = (uint4*)sWt;
    for (int i = tid; i < (64 * S) / 8; i += TPB) dst[i] = src[i];
  }
  __syncthreads();

  int wv   = tid >> 5;
  int lane = tid & 31;
  int m0   = (blockIdx.x * 2 + (wv >> 2)) * 16;   // row tile base
  int col  = ((wv & 3) << 4) + (lane & 15);       // output column (0..63)

  int row  = m0 + (lane & 15);
  int rowc = row < nrows ? row : (nrows - 1);     // clamp pad rows (store guarded)

  int half = (lane < 16) ? 0 : 8;                 // ISA 7.12.2 K-half per lane group
  const __bf16* arow = A   + (size_t)rowc * K + half;
  const __bf16* brow = sWt + col * S + half;

  v8f acc = {};
#pragma unroll
  for (int k0 = 0; k0 < K; k0 += 32) {
    // fragment elems 0..7 -> K = base+0..7, elems 8..15 -> K = base+16..23
    v8bf alo = *(const v8bf*)(arow + k0);
    v8bf ahi = *(const v8bf*)(arow + k0 + 16);
    v8bf blo = *(const v8bf*)(brow + k0);
    v8bf bhi = *(const v8bf*)(brow + k0 + 16);
    v16bf a = __builtin_shufflevector(alo, ahi, 0, 1, 2, 3, 4, 5, 6, 7,
                                      8, 9, 10, 11, 12, 13, 14, 15);
    v16bf b = __builtin_shufflevector(blo, bhi, 0, 1, 2, 3, 4, 5, 6, 7,
                                      8, 9, 10, 11, 12, 13, 14, 15);
    acc = __builtin_amdgcn_wmma_f32_16x16x32_bf16(false, a, false, b, (short)0,
                                                  acc, false, false);
  }

  // D layout: c[j] -> M = m0 + j + (lane>=16 ? 8 : 0), N = col
  int mb = m0 + half;
  float* cp = Cout + (size_t)mb * 64 + col;
  if (m0 + 16 <= nrows) {
    // wave-uniform fast path: full tile, unguarded stores
#pragma unroll
    for (int j = 0; j < 8; ++j) cp[(size_t)j * 64] = acc[j];
  } else {
    // boundary tile (last block only)
#pragma unroll
    for (int j = 0; j < 8; ++j) {
      if (mb + j < nrows) cp[(size_t)j * 64] = acc[j];
    }
  }
}

// ---------------- per-node attention projections ----------------
__global__ void k_alpha(const float* __restrict__ h, const float* __restrict__ a_s,
                        const float* __restrict__ a_d, float* __restrict__ als,
                        float* __restrict__ ald, int Nn, int H, int C) {
  int i = blockIdx.x * blockDim.x + threadIdx.x;
  if (i >= Nn * H) return;
  int n = i / H, hd = i - n * H;
  const float* hp = h + (size_t)n * 64 + hd * C;
  float s1 = 0.f, s2 = 0.f;
  for (int c = 0; c < C; ++c) {
    float v = hp[c];
    s1 += v * a_s[hd * C + c];
    s2 += v * a_d[hd * C + c];
  }
  als[i] = s1;
  ald[i] = s2;
}

// ---------------- per-layer state init (fresh every call) ----------------
__global__ void k_init(float* __restrict__ agg, float* __restrict__ mx,
                       float* __restrict__ sm, int Nn, int H) {
  int i = blockIdx.x * blockDim.x + threadIdx.x;
  if (i < Nn * 64) agg[i] = 0.f;
  if (i < Nn * H) { mx[i] = -3.0e38f; sm[i] = 0.f; }
}

__device__ inline void atomicMaxF(float* a, float v) {
  int* ai = (int*)a;
  int cur = *ai;
  while (__int_as_float(cur) < v) {
    int prev = atomicCAS(ai, cur, __float_as_int(v));
    if (prev == cur) break;
    cur = prev;
  }
}

// ---------------- edge pass A: logits + segment max ----------------
__global__ void k_edge_logit(const long long* __restrict__ ei,
                             const float* __restrict__ als, const float* __restrict__ ald,
                             float* __restrict__ elg, float* __restrict__ mx,
                             int E, int Etot, int H) {
  int i = blockIdx.x * blockDim.x + threadIdx.x;
  if (i >= Etot * H) return;
  int e = i / H, hd = i - e * H;
  long long s, d;
  if (e < E) { s = ei[e]; d = ei[E + e]; } else { s = d = (long long)(e - E); }
  float a = als[s * H + hd] + ald[d * H + hd];
  float lr = a > 0.f ? a : 0.2f * a;              // leaky_relu 0.2
  elg[i] = lr;
  atomicMaxF(&mx[d * H + hd], lr);
}

// ---------------- edge pass B: exp + segment sum ----------------
__global__ void k_edge_exp(const long long* __restrict__ ei, float* __restrict__ elg,
                           const float* __restrict__ mx, float* __restrict__ sm,
                           int E, int Etot, int H) {
  int i = blockIdx.x * blockDim.x + threadIdx.x;
  if (i >= Etot * H) return;
  int e = i / H, hd = i - e * H;
  long long d;
  if (e < E) { d = ei[E + e]; } else { d = (long long)(e - E); }
  float ex = __expf(elg[i] - mx[d * H + hd]);
  elg[i] = ex;
  atomicAdd(&sm[d * H + hd], ex);
}

// ---------------- edge pass C: weighted scatter-add (wave per edge) ----------------
__global__ void k_edge_msg(const long long* __restrict__ ei, const float* __restrict__ h,
                           const float* __restrict__ elg, const float* __restrict__ sm,
                           float* __restrict__ agg, int E, int Etot, int H, int C) {
  int gid  = blockIdx.x * 8 + (threadIdx.x >> 5);
  int lane = threadIdx.x & 31;
  if (gid >= Etot) return;
  long long s, d;
  if (gid < E) { s = ei[gid]; d = ei[E + gid]; } else { s = d = (long long)(gid - E); }
  for (int c = lane; c < 64; c += 32) {
    int hd = c / C;
    float w = elg[(size_t)gid * H + hd] / (sm[d * H + hd] + 1e-16f);
    atomicAdd(&agg[d * 64 + c], h[s * 64 + c] * w);
  }
}

// ---------------- bias + leaky(0.01) -> f32 activation + bf16 next input ----------------
__global__ void k_post(const float* __restrict__ agg, const float* __restrict__ b,
                       float* __restrict__ act, __bf16* __restrict__ xin, int Nn) {
  int i = blockIdx.x * blockDim.x + threadIdx.x;
  if (i >= Nn * 64) return;
  int f = i & 63;
  float v = agg[i] + b[f];
  float o = v > 0.f ? v : 0.01f * v;
  act[i] = o;
  xin[i] = (__bf16)o;
}

// ---------------- global mean pool ----------------
__global__ void k_pool_init(float* __restrict__ psum, float* __restrict__ pcnt, int G) {
  int i = blockIdx.x * blockDim.x + threadIdx.x;
  if (i < G * 64) psum[i] = 0.f;
  if (i < G) pcnt[i] = 0.f;
}

__global__ void k_pool_acc(const float* __restrict__ act, const long long* __restrict__ batch,
                           float* __restrict__ psum, float* __restrict__ pcnt, int Nn) {
  int i = blockIdx.x * blockDim.x + threadIdx.x;
  if (i >= Nn * 64) return;
  int n = i >> 6, f = i & 63;
  int g = (int)batch[n];
  atomicAdd(&psum[g * 64 + f], act[i]);
  if (f == 0) atomicAdd(&pcnt[g], 1.0f);
}

__global__ void k_pool_fin(const float* __restrict__ psum, const float* __restrict__ pcnt,
                           float* __restrict__ out, int G) {
  int i = blockIdx.x * blockDim.x + threadIdx.x;
  if (i >= G * 64) return;
  out[i] = psum[i] / fmaxf(pcnt[i >> 6], 1.0f);
}

// ---------------------------------------------------------------------------

static inline int cdiv(int a, int b) { return (a + b - 1) / b; }

extern "C" void kernel_launch(void* const* d_in, const int* in_sizes, int n_in,
                              void* d_out, int out_size, void* d_ws, size_t ws_size,
                              hipStream_t stream) {
  const int N = in_sizes[0] / 128;     // 50000
  const int E = in_sizes[1] / 2;       // 1600000
  const int Etot = E + N;              // with self-loops
  const int G = out_size / 64;         // 64

  const float*     x     = (const float*)d_in[0];
  const long long* ei    = (const long long*)d_in[1];
  const long long* batch = (const long long*)d_in[2];
  const float* Wp[4]  = { (const float*)d_in[3],  (const float*)d_in[7],
                          (const float*)d_in[11], (const float*)d_in[15] };
  const float* asp[4] = { (const float*)d_in[4],  (const float*)d_in[8],
                          (const float*)d_in[12], (const float*)d_in[16] };
  const float* adp[4] = { (const float*)d_in[5],  (const float*)d_in[9],
                          (const float*)d_in[13], (const float*)d_in[17] };
  const float* bp[4]  = { (const float*)d_in[6],  (const float*)d_in[10],
                          (const float*)d_in[14], (const float*)d_in[18] };

  const int Fin[4] = {128, 64, 64, 64};
  const int Hh[4]  = {4, 4, 4, 1};
  const int Cc[4]  = {16, 16, 16, 64};

  // workspace layout (256B aligned slices)
  char* ws = (char*)d_ws;
  size_t off = 0;
  auto alloc = [&](size_t bytes) -> char* {
    char* p = ws + off;
    off = (off + bytes + 255) & ~(size_t)255;
    return p;
  };
  __bf16* bf_in = (__bf16*)alloc((size_t)N * 128 * 2);
  __bf16* bf_wT = (__bf16*)alloc((size_t)64 * 136 * 2);   // max stride 128+8
  float*  h     = (float*)alloc((size_t)N * 64 * 4);
  float*  als   = (float*)alloc((size_t)N * 4 * 4);
  float*  ald   = (float*)alloc((size_t)N * 4 * 4);
  float*  mx    = (float*)alloc((size_t)N * 4 * 4);
  float*  sm    = (float*)alloc((size_t)N * 4 * 4);
  float*  elg   = (float*)alloc((size_t)Etot * 4 * 4);
  float*  agg   = (float*)alloc((size_t)N * 64 * 4);
  float*  act   = (float*)alloc((size_t)N * 64 * 4);
  float*  psum  = (float*)alloc((size_t)G * 64 * 4);
  float*  pcnt  = (float*)alloc((size_t)G * 4);
  (void)ws_size;

  // layer 1 input -> bf16
  k_cvt_bf16<<<cdiv(N * 128, TPB), TPB, 0, stream>>>(x, bf_in, N * 128);

  for (int l = 0; l < 4; ++l) {
    const int K = Fin[l], H = Hh[l], C = Cc[l];
    const int S = K + 8;

    k_cvt_wT<<<cdiv(K * 64, TPB), TPB, 0, stream>>>(Wp[l], bf_wT, K, S);
    if (K == 128)
      k_gemm_wmma<128><<<cdiv(N, 32), TPB, 0, stream>>>(bf_in, bf_wT, h, N);
    else
      k_gemm_wmma<64><<<cdiv(N, 32), TPB, 0, stream>>>(bf_in, bf_wT, h, N);
    k_alpha<<<cdiv(N * H, TPB), TPB, 0, stream>>>(h, asp[l], adp[l], als, ald, N, H, C);
    k_init<<<cdiv(N * 64, TPB), TPB, 0, stream>>>(agg, mx, sm, N, H);
    k_edge_logit<<<cdiv(Etot * H, TPB), TPB, 0, stream>>>(ei, als, ald, elg, mx, E, Etot, H);
    k_edge_exp<<<cdiv(Etot * H, TPB), TPB, 0, stream>>>(ei, elg, mx, sm, E, Etot, H);
    k_edge_msg<<<cdiv(Etot, 8), TPB, 0, stream>>>(ei, h, elg, sm, agg, E, Etot, H, C);
    k_post<<<cdiv(N * 64, TPB), TPB, 0, stream>>>(agg, bp[l], act, bf_in, N);
  }

  k_pool_init<<<cdiv(G * 64, TPB), TPB, 0, stream>>>(psum, pcnt, G);
  k_pool_acc<<<cdiv(N * 64, TPB), TPB, 0, stream>>>(act, batch, psum, pcnt, N);
  k_pool_fin<<<cdiv(G * 64, TPB), TPB, 0, stream>>>(psum, pcnt, (float*)d_out, G);
}
